// MultiResAttention_72919954751806
// MI455X (gfx1250) — matrix operand order repaired and checked by reading, not compiled, not run on hardware
//
#include <hip/hip_runtime.h>
#include <math.h>

// ---------------------------------------------------------------------------
// MultiResAttention for MI455X (gfx1250): bf16 WMMA (16x16x32) everywhere,
// f32 accumulate. wave32. All matmul operands staged so fragments load
// contiguous rows. Native bf16 converts (v_cvt) instead of bit-twiddling.
// ---------------------------------------------------------------------------

typedef __attribute__((ext_vector_type(16))) __bf16 v16bf;
typedef __attribute__((ext_vector_type(8)))  float  v8f;

#define NN   32768
#define DIMD 256
#define HH   8
#define DD   32
#define PP   128
#define SS   256
#define MM   4
#define RR   512           // P*M
#define INV_SCALE 0.17677669529663688f   // 1/sqrt(32)

__device__ __forceinline__ __bf16 f2bf(float f) { return (__bf16)f; }
__device__ __forceinline__ float  bf2f(__bf16 b) { return (float)b; }

__device__ __forceinline__ v8f wmma_bf16(v16bf a, v16bf b, v8f c) {
  return __builtin_amdgcn_wmma_f32_16x16x32_bf16(false, a, false, b,
                                                 (short)0, c, false, false);
}

// A fragment (16x32 bf16, M x K): lane holds row (rowBase + lane%16).
// ISA layout: half0 -> V0-3:K0-7, V4-7:K16-23 ; half1 -> K8-15, K24-31.
__device__ __forceinline__ v16bf load_a_bf(const __bf16* base, int ld,
                                           int row, int kb) {
  int lane = threadIdx.x & 31;
  const __bf16* p = base + (size_t)(row + (lane & 15)) * ld + kb
                         + ((lane >> 4) << 3);
  v16bf f;
#pragma unroll
  for (int i = 0; i < 8; ++i) { f[i] = p[i]; f[i + 8] = p[16 + i]; }
  return f;
}

// A fragment from fp32 source (convert on load, native v_cvt).
__device__ __forceinline__ v16bf load_a_f32(const float* base, int ld,
                                            int row, int kb) {
  int lane = threadIdx.x & 31;
  const float* p = base + (size_t)(row + (lane & 15)) * ld + kb
                        + ((lane >> 4) << 3);
  v16bf f;
#pragma unroll
  for (int i = 0; i < 8; ++i) { f[i] = f2bf(p[i]); f[i + 8] = f2bf(p[16 + i]); }
  return f;
}

// B fragment (32x16 bf16, K x N). Source stored column-contiguous: per-lane
// column N = colBase + lane%16, K = kb + half*8 + e (e<8), kb + 16 + half*8 + (e-8) (e>=8).
__device__ __forceinline__ v16bf load_b_bf(const __bf16* base, int ld,
                                           int col, int kb) {
  int lane = threadIdx.x & 31;
  const __bf16* p = base + (size_t)(col + (lane & 15)) * ld + kb
                         + ((lane >> 4) << 3);
  v16bf f;
#pragma unroll
  for (int i = 0; i < 8; ++i) { f[i] = p[i]; f[i + 8] = p[16 + i]; }
  return f;
}

// ---------------------------------------------------------------------------
// Kernel 1: transpose+convert the 4 weight matrices to bf16 (WT[j][k]=W[k][j])
// ---------------------------------------------------------------------------
__global__ void __launch_bounds__(256)
transpose_w_kernel(const float* wq, const float* wk, const float* wv,
                   const float* wo, __bf16* WT) {
  int mat = blockIdx.x >> 8;
  int j   = blockIdx.x & 255;
  int k   = threadIdx.x;
  const float* W = (mat == 0) ? wq : (mat == 1) ? wk : (mat == 2) ? wv : wo;
  WT[(size_t)mat * DIMD * DIMD + (size_t)j * DIMD + k] = f2bf(W[(size_t)k * DIMD + j]);
}

// ---------------------------------------------------------------------------
// Kernel 2: Q/K/V = bf16(x @ W + b).  Block = 8 waves, 16 rows x 256 cols.
// ---------------------------------------------------------------------------
__global__ void __launch_bounds__(256)
qkv_kernel(const float* x, const __bf16* WT,
           const float* bq, const float* bk, const float* bv,
           __bf16* Qbf, __bf16* Kbf, __bf16* Vbf) {
  int lane = threadIdx.x & 31, wv_ = threadIdx.x >> 5;
  int n = lane & 15, half = lane >> 4;
  int rbase = blockIdx.x * 16, cbase = wv_ * 32;

  v16bf a[8];
#pragma unroll
  for (int kc = 0; kc < 8; ++kc) a[kc] = load_a_f32(x, DIMD, rbase, kc * 32);

#pragma unroll
  for (int o = 0; o < 3; ++o) {
    const __bf16* Wt = WT + (size_t)o * DIMD * DIMD;
    const float*  bias = (o == 0) ? bq : (o == 1) ? bk : bv;
    __bf16* out = (o == 0) ? Qbf : (o == 1) ? Kbf : Vbf;
#pragma unroll
    for (int nb = 0; nb < 2; ++nb) {
      v8f acc = {};
#pragma unroll
      for (int kc = 0; kc < 8; ++kc)
        acc = wmma_bf16(a[kc], load_b_bf(Wt, DIMD, cbase + nb * 16, kc * 32), acc);
      int col = cbase + nb * 16 + n;
      float bb = bias[col];
#pragma unroll
      for (int j = 0; j < 8; ++j)
        out[(size_t)(rbase + half * 8 + j) * DIMD + col] = f2bf(acc[j] + bb);
    }
  }
}

// ---------------------------------------------------------------------------
// Kernel 3: gate alpha[n] = sigmoid(x[n]·gw + gb)
// ---------------------------------------------------------------------------
__global__ void __launch_bounds__(256)
gate_kernel(const float* x, const float* gw, const float* gb, float* alpha) {
  int nidx = blockIdx.x * 256 + threadIdx.x;
  const float* xr = x + (size_t)nidx * DIMD;
  float s = 0.f;
#pragma unroll 8
  for (int k = 0; k < DIMD; ++k) s = fmaf(xr[k], gw[k], s);
  s += gb[0];
  alpha[nidx] = 1.f / (1.f + expf(-s));
}

// ---------------------------------------------------------------------------
// Kernel 4: gather per-partition operands.
//   Qp,Kp : [P][H][S][D]  row-contiguous in D
//   Vt    : [P][H][D][S]  row-contiguous in S (V transposed)
// ---------------------------------------------------------------------------
__global__ void __launch_bounds__(256)
stage_kernel(const int* idx, const __bf16* Qbf, const __bf16* Kbf,
             const __bf16* Vbf, __bf16* Qp, __bf16* Kp, __bf16* Vt) {
  int gid = blockIdx.x * 256 + threadIdx.x;
  int d = gid & 31, h = (gid >> 5) & 7, s = (gid >> 8) & 255, p = gid >> 16;
  int node = idx[p * SS + s];
  size_t src = (size_t)node * DIMD + h * DD + d;
  size_t ph  = (size_t)(p * HH + h);
  Qp[(ph * SS + s) * DD + d] = Qbf[src];
  Kp[(ph * SS + s) * DD + d] = Kbf[src];
  Vt[(ph * DD + d) * SS + s] = Vbf[src];
}

// ---------------------------------------------------------------------------
// Kernel 5: local attention. 1 wave per block; each wave: 16 query rows of one
// (p,h): scores 16x256 in regs -> softmax -> probs@Vp -> scatter to out_local.
// ---------------------------------------------------------------------------
__global__ void __launch_bounds__(32)
local_attn_kernel(const __bf16* Qp, const __bf16* Kp, const __bf16* Vt,
                  const int* idx, float* out_local) {
  __shared__ __bf16 probs[16 * SS];   // 8 KB
  int lane = threadIdx.x & 31;
  int n = lane & 15, half = lane >> 4;
  int t16 = blockIdx.x & 15;
  int ph  = blockIdx.x >> 4;
  int h = ph & 7, p = ph >> 3;
  const __bf16* Qph = Qp + (size_t)ph * SS * DD;
  const __bf16* Kph = Kp + (size_t)ph * SS * DD;
  const __bf16* Vph = Vt + (size_t)ph * DD * SS;
  int srow = t16 * 16;

  v16bf qa = load_a_bf(Qph, DD, srow, 0);
  v8f acc[16];
#pragma unroll
  for (int t = 0; t < 16; ++t) {
    v8f z = {};
    acc[t] = wmma_bf16(qa, load_b_bf(Kph, DD, t * 16, 0), z);
  }

  float rmax[8], rinv[8];
#pragma unroll
  for (int j = 0; j < 8; ++j) {
    float mx = -3.4e38f;
#pragma unroll
    for (int t = 0; t < 16; ++t) mx = fmaxf(mx, acc[t][j]);
#pragma unroll
    for (int msk = 1; msk < 16; msk <<= 1)
      mx = fmaxf(mx, __shfl_xor(mx, msk, 32));
    rmax[j] = mx * INV_SCALE;
    float sum = 0.f;
#pragma unroll
    for (int t = 0; t < 16; ++t) {
      float e = expf(fmaf(acc[t][j], INV_SCALE, -rmax[j]));
      acc[t][j] = e;
      sum += e;
    }
#pragma unroll
    for (int msk = 1; msk < 16; msk <<= 1)
      sum += __shfl_xor(sum, msk, 32);
    rinv[j] = 1.f / sum;
  }
#pragma unroll
  for (int t = 0; t < 16; ++t)
#pragma unroll
    for (int j = 0; j < 8; ++j)
      probs[(half * 8 + j) * SS + t * 16 + n] = f2bf(acc[t][j] * rinv[j]);
  __syncthreads();

#pragma unroll
  for (int nb = 0; nb < 2; ++nb) {
    v8f o = {};
#pragma unroll
    for (int kc = 0; kc < 8; ++kc)
      o = wmma_bf16(load_a_bf(probs, SS, 0, kc * 32),
                    load_b_bf(Vph, SS, nb * 16, kc * 32), o);
#pragma unroll
    for (int j = 0; j < 8; ++j) {
      int s = srow + half * 8 + j;
      int node = idx[p * SS + s];
      out_local[(size_t)node * DIMD + h * DD + nb * 16 + n] = o[j];
    }
  }
}

// ---------------------------------------------------------------------------
// Kernel 6: pooled representatives per (p,h). Small; VALU + LDS reductions.
//   reps_k : [R][H*D] bf16, reps_vT : [H][D][R] bf16
// ---------------------------------------------------------------------------
__global__ void __launch_bounds__(256)
pool_kernel(const float* seeds, const __bf16* Kp, const __bf16* Vt,
            __bf16* reps_k, __bf16* reps_vT) {
  __shared__ float buf[256];
  __shared__ float pr[MM][SS];
  int tid = threadIdx.x;
  int ph = blockIdx.x, h = ph & 7, p = ph >> 3;
  const __bf16* Kph = Kp + (size_t)ph * SS * DD;
  const __bf16* Vph = Vt + (size_t)ph * DD * SS;

  float sc[MM];
  {
    int s = tid;
#pragma unroll
    for (int m = 0; m < MM; ++m) {
      float acc = 0.f;
#pragma unroll
      for (int d = 0; d < DD; ++d)
        acc = fmaf(seeds[(m * HH + h) * DD + d], bf2f(Kph[s * DD + d]), acc);
      sc[m] = acc * INV_SCALE;
    }
  }
#pragma unroll
  for (int m = 0; m < MM; ++m) {
    buf[tid] = sc[m]; __syncthreads();
    for (int o = 128; o > 0; o >>= 1) {
      if (tid < o) buf[tid] = fmaxf(buf[tid], buf[tid + o]);
      __syncthreads();
    }
    float mx = buf[0]; __syncthreads();
    float e = expf(sc[m] - mx);
    buf[tid] = e; __syncthreads();
    for (int o = 128; o > 0; o >>= 1) {
      if (tid < o) buf[tid] += buf[tid + o];
      __syncthreads();
    }
    float sum = buf[0]; __syncthreads();
    pr[m][tid] = e / sum;
  }
  __syncthreads();

  if (tid < 128) {             // reps_k
    int m = tid >> 5, d = tid & 31;
    float acc = 0.f;
    for (int s = 0; s < SS; ++s) acc = fmaf(pr[m][s], bf2f(Kph[s * DD + d]), acc);
    reps_k[(size_t)(p * MM + m) * DIMD + h * DD + d] = f2bf(acc);
  } else {                     // reps_v (transposed per head)
    int t = tid - 128;
    int m = t >> 5, d = t & 31;
    float acc = 0.f;
    for (int s = 0; s < SS; ++s) acc = fmaf(pr[m][s], bf2f(Vph[d * SS + s]), acc);
    reps_vT[((size_t)h * DD + d) * RR + (p * MM + m)] = f2bf(acc);
  }
}

// ---------------------------------------------------------------------------
// Kernel 7: global cross-attention. 1 wave per (16 nodes, head).
// ---------------------------------------------------------------------------
__global__ void __launch_bounds__(32)
cross_attn_kernel(const __bf16* Qbf, const __bf16* reps_k,
                  const __bf16* reps_vT, float* out_global) {
  __shared__ float  sc[16 * RR];       // 32 KB
  __shared__ __bf16 pb[16 * RR];       // 16 KB
  int lane = threadIdx.x & 31;
  int n = lane & 15, half = lane >> 4;
  int h = blockIdx.x & 7;
  int nbase = (blockIdx.x >> 3) * 16;

  v16bf qa = load_a_bf(Qbf + h * DD, DIMD, nbase, 0);
#pragma unroll
  for (int t = 0; t < 32; ++t) {
    v8f z = {};
    v8f c = wmma_bf16(qa, load_b_bf(reps_k + h * DD, DIMD, t * 16, 0), z);
#pragma unroll
    for (int j = 0; j < 8; ++j)
      sc[(half * 8 + j) * RR + t * 16 + n] = c[j];
  }
  __syncthreads();

  // softmax: 2 lanes per row (halves split the 512 columns)
  {
    int r = lane & 15, cb = half * 256;
    float mx = -3.4e38f;
    for (int c = 0; c < 256; ++c) mx = fmaxf(mx, sc[r * RR + cb + c]);
    mx = fmaxf(mx, __shfl_xor(mx, 16, 32));
    mx *= INV_SCALE;
    float sum = 0.f;
    for (int c = 0; c < 256; ++c) {
      float e = expf(fmaf(sc[r * RR + cb + c], INV_SCALE, -mx));
      sc[r * RR + cb + c] = e;
      sum += e;
    }
    sum += __shfl_xor(sum, 16, 32);
    float rs = 1.f / sum;
    for (int c = 0; c < 256; ++c)
      pb[r * RR + cb + c] = f2bf(sc[r * RR + cb + c] * rs);
  }
  __syncthreads();

#pragma unroll
  for (int nb = 0; nb < 2; ++nb) {
    v8f o = {};
#pragma unroll
    for (int kc = 0; kc < 16; ++kc)
      o = wmma_bf16(load_a_bf(pb, RR, 0, kc * 32),
                    load_b_bf(reps_vT + (size_t)h * DD * RR, RR, nb * 16, kc * 32), o);
#pragma unroll
    for (int j = 0; j < 8; ++j)
      out_global[(size_t)(nbase + half * 8 + j) * DIMD + h * DD + nb * 16 + n] = o[j];
  }
}

// ---------------------------------------------------------------------------
// Kernel 8: h = alpha*loc + (1-alpha)*glob ; out = h @ Wo + bo
// ---------------------------------------------------------------------------
__global__ void __launch_bounds__(256)
final_kernel(const float* loc, const float* glob, const float* alpha,
             const __bf16* WoT, const float* bo, float* out) {
  int lane = threadIdx.x & 31, wv_ = threadIdx.x >> 5;
  int n = lane & 15, half = lane >> 4;
  int rbase = blockIdx.x * 16, cbase = wv_ * 32;
  int r = rbase + (lane & 15);
  float al = alpha[r];

  v16bf a[8];
#pragma unroll
  for (int kc = 0; kc < 8; ++kc) {
    const float* pl = loc  + (size_t)r * DIMD + kc * 32 + half * 8;
    const float* pg = glob + (size_t)r * DIMD + kc * 32 + half * 8;
    v16bf f;
#pragma unroll
    for (int i = 0; i < 8; ++i) {
      f[i]     = f2bf(fmaf(al, pl[i]      - pg[i],      pg[i]));
      f[i + 8] = f2bf(fmaf(al, pl[16 + i] - pg[16 + i], pg[16 + i]));
    }
    a[kc] = f;
  }
#pragma unroll
  for (int nb = 0; nb < 2; ++nb) {
    v8f acc = {};
#pragma unroll
    for (int kc = 0; kc < 8; ++kc)
      acc = wmma_bf16(a[kc], load_b_bf(WoT, DIMD, cbase + nb * 16, kc * 32), acc);
    int col = cbase + nb * 16 + n;
    float bb = bo[col];
#pragma unroll
    for (int j = 0; j < 8; ++j)
      out[(size_t)(rbase + half * 8 + j) * DIMD + col] = acc[j] + bb;
  }
}

// ---------------------------------------------------------------------------
extern "C" void kernel_launch(void* const* d_in, const int* in_sizes, int n_in,
                              void* d_out, int out_size, void* d_ws, size_t ws_size,
                              hipStream_t stream) {
  const float* x    = (const float*)d_in[0];
  const int*   idx  = (const int*)d_in[1];
  const float* Wq   = (const float*)d_in[2];
  const float* bq   = (const float*)d_in[3];
  const float* Wk   = (const float*)d_in[4];
  const float* bk   = (const float*)d_in[5];
  const float* Wv   = (const float*)d_in[6];
  const float* bv   = (const float*)d_in[7];
  const float* Wo   = (const float*)d_in[8];
  const float* bo   = (const float*)d_in[9];
  const float* seeds= (const float*)d_in[10];
  const float* gw   = (const float*)d_in[11];
  const float* gb   = (const float*)d_in[12];
  float* out = (float*)d_out;

  char* w = (char*)d_ws;
  size_t off = 0;
  auto take = [&](size_t bytes) -> void* {
    void* p = w + off;
    off += (bytes + 255) & ~(size_t)255;
    return p;
  };
  __bf16* Qbf   = (__bf16*)take((size_t)NN * DIMD * 2);
  __bf16* Kbf   = (__bf16*)take((size_t)NN * DIMD * 2);
  __bf16* Vbf   = (__bf16*)take((size_t)NN * DIMD * 2);
  __bf16* Qp    = (__bf16*)take((size_t)PP * HH * SS * DD * 2);
  __bf16* Kp    = (__bf16*)take((size_t)PP * HH * SS * DD * 2);
  __bf16* Vt    = (__bf16*)take((size_t)PP * HH * DD * SS * 2);
  __bf16* WT    = (__bf16*)take((size_t)4 * DIMD * DIMD * 2);
  float*  o_loc = (float*)take((size_t)NN * DIMD * 4);
  float*  o_glb = (float*)take((size_t)NN * DIMD * 4);
  float*  alpha = (float*)take((size_t)NN * 4);
  __bf16* rk    = (__bf16*)take((size_t)RR * DIMD * 2);
  __bf16* rvT   = (__bf16*)take((size_t)HH * DD * RR * 2);

  transpose_w_kernel<<<4 * 256, 256, 0, stream>>>(Wq, Wk, Wv, Wo, WT);
  qkv_kernel<<<NN / 16, 256, 0, stream>>>(x, WT, bq, bk, bv, Qbf, Kbf, Vbf);
  gate_kernel<<<NN / 256, 256, 0, stream>>>(x, gw, gb, alpha);
  stage_kernel<<<(NN * DIMD) / 256, 256, 0, stream>>>(idx, Qbf, Kbf, Vbf, Qp, Kp, Vt);
  local_attn_kernel<<<PP * HH * (SS / 16), 32, 0, stream>>>(Qp, Kp, Vt, idx, o_loc);
  pool_kernel<<<PP * HH, 256, 0, stream>>>(seeds, Kp, Vt, rk, rvT);
  cross_attn_kernel<<<(NN / 16) * HH, 32, 0, stream>>>(Qbf, rk, rvT, o_glb);
  final_kernel<<<NN / 16, 256, 0, stream>>>(o_loc, o_glb, alpha,
                                            WT + (size_t)3 * DIMD * DIMD, bo, out);
}
